// ProtienGAT_68367289418035
// MI455X (gfx1250) — matrix operand order repaired and checked by reading, not compiled, not run on hardware
//
#include <hip/hip_runtime.h>

#define Nn 8192
#define Kk 32
#define Hh 128
#define Vv 21
#define NK (Nn*Kk)
#define NEGS 0.2f

typedef __attribute__((ext_vector_type(16))) __bf16 v16bf;
typedef __attribute__((ext_vector_type(8)))  float  v8f;

static __device__ __forceinline__ unsigned short bf_bits(__bf16 b){ union{ __bf16 b; unsigned short u; } x; x.b=b; return x.u; }
static __device__ __forceinline__ __bf16 bits_bf(unsigned short u){ union{ unsigned short u; __bf16 b; } x; x.u=u; return x.b; }
// monotonic float<->u32 encoding for atomic max
static __device__ __forceinline__ unsigned fenc(float f){ unsigned u=__float_as_uint(f); return (u&0x80000000u)? ~u : (u|0x80000000u); }
static __device__ __forceinline__ float fdec(unsigned u){ return (u&0x80000000u)? __uint_as_float(u&0x7fffffffu) : __uint_as_float(~u); }

// ---------------- small prep kernels ----------------

__global__ void k_keys(const float* __restrict__ chain_M, const float* __restrict__ mask,
                       const float* __restrict__ noise, float* __restrict__ keys) {
  int i = blockIdx.x*blockDim.x + threadIdx.x;
  if (i < Nn) keys[i] = (chain_M[i]*mask[i] + 0.0001f) * fabsf(noise[i]);
}

// inv[i] = rank of i under stable ascending argsort of keys (argsort(argsort(key)))
__global__ void k_rank(const float* __restrict__ keys, int* __restrict__ inv) {
  int i = blockIdx.x*blockDim.x + threadIdx.x;
  if (i >= Nn) return;
  float ki = keys[i];
  int r = 0;
  for (int j = 0; j < Nn; ++j) {
    float kj = keys[j];
    r += (kj < ki) || (kj == ki && j < i);
  }
  inv[i] = r;
}

__global__ void k_init(unsigned* __restrict__ m2, float* __restrict__ d2,
                       float* __restrict__ SA, float* __restrict__ SB, int* __restrict__ occ) {
  int i = blockIdx.x*blockDim.x + threadIdx.x;
  if (i < Nn) { m2[i] = 0u; d2[i] = 0.f; SA[i] = 0.f; SB[i] = 0.f; occ[i] = 0; }
}

__global__ void k_occ(const int* __restrict__ E_idx, int* __restrict__ occ) {
  int e = blockIdx.x*blockDim.x + threadIdx.x;
  if (e < NK) occ[E_idx[e]] = 1;
}

// Precompute the collapsed per-node vectors and tables (tiny: one block).
// node in {g1_b, g1_b+g1_bl}  ->  xl2/xr2 have two variants each.
// P2[s][n] = sum_k W_s[s][k] * g2_We[H+k][n]
// pA/pB = xl2A/B @ W_out ; pbase = g2_b @ W_out + b_out
__global__ __launch_bounds__(256) void k_prep(
    const float* __restrict__ W_s, const float* __restrict__ g1_bl, const float* __restrict__ g1_b,
    const float* __restrict__ g2_Wl, const float* __restrict__ g2_bl,
    const float* __restrict__ g2_Wr, const float* __restrict__ g2_br,
    const float* __restrict__ g2_We, const float* __restrict__ g2_b,
    const float* __restrict__ W_out, const float* __restrict__ b_out,
    float* __restrict__ xl2A, float* __restrict__ xl2B,
    float* __restrict__ xr2A, float* __restrict__ xr2B,
    float* __restrict__ P2, float* __restrict__ pA, float* __restrict__ pB, float* __restrict__ pbase) {
  int t = threadIdx.x;
  if (t < Hh) {
    int n = t;
    float a1 = g2_bl[n], b1 = g2_bl[n], a2 = g2_br[n], b2 = g2_br[n];
    for (int k = 0; k < Hh; ++k) {
      float nA = g1_b[k];
      float nB = nA + g1_bl[k];
      float wl = g2_Wl[k*Hh + n], wr = g2_Wr[k*Hh + n];
      a1 += nA*wl; b1 += nB*wl; a2 += nA*wr; b2 += nB*wr;
    }
    xl2A[n] = a1; xl2B[n] = b1; xr2A[n] = a2; xr2B[n] = b2;
  }
  for (int idx = t; idx < Vv*Hh; idx += 256) {
    int s = idx / Hh, n = idx % Hh;
    float acc = 0.f;
    for (int k = 0; k < Hh; ++k) acc += W_s[s*Hh + k] * g2_We[(Hh + k)*Hh + n];
    P2[idx] = acc;
  }
  __syncthreads();
  if (t < Vv) {
    int v = t;
    float a = 0.f, b = 0.f, c = b_out[v];
    for (int n = 0; n < Hh; ++n) {
      float w = W_out[n*Vv + v];
      a += xl2A[n]*w; b += xl2B[n]*w; c += g2_b[n]*w;
    }
    pA[v] = a; pB[v] = b; pbase[v] = c;
  }
}

// ---------------- big WMMA kernel 1: h_e = E @ W_e + b_e (bf16 out) ----------------
// 8 waves/block; each wave: one 16-edge tile, full H=128 output, K-loop of 4 x (16x16x32 bf16 WMMA).
__global__ __launch_bounds__(256) void k_he_gemm(
    const float* __restrict__ E, const float* __restrict__ W_e, const float* __restrict__ b_e,
    unsigned short* __restrict__ he) {
  __shared__ __align__(32) __bf16 Wsh[Hh*Hh];   // transposed [n][k]
  __shared__ float besh[Hh];
  int tid = threadIdx.x;
  for (int idx = tid; idx < Hh*Hh; idx += 256) {
    int k = idx / Hh, n = idx % Hh;
    Wsh[n*Hh + k] = (__bf16)W_e[idx];
  }
  if (tid < Hh) besh[tid] = b_e[tid];
  __syncthreads();

  int wave = tid >> 5, lane = tid & 31;
  int hi = lane >> 4, lo = lane & 15;
  size_t eb = ((size_t)blockIdx.x * 8 + wave) * 16;
  const float* Erow = E + (eb + lo) * Hh;   // A row m = lo

  v8f c[8] = {};
  for (int kt = 0; kt < 4; ++kt) {
    v16bf a;
    #pragma unroll
    for (int r = 0; r < 8; ++r) {
      int e = 2*r;
      int k = 16*hi + e;   // A: k = 16*hi + e contiguous (matches B)
      float2 f = *(const float2*)(Erow + kt*32 + k);
      a[e]   = (__bf16)f.x;
      a[e+1] = (__bf16)f.y;
    }
    #pragma unroll
    for (int nt = 0; nt < 8; ++nt) {
      int n = nt*16 + lo;
      v16bf b = *(const v16bf*)&Wsh[n*Hh + kt*32 + hi*16];  // B: k = 16*hi + e contiguous
      c[nt] = __builtin_amdgcn_wmma_f32_16x16x32_bf16(false, a, false, b, (short)0, c[nt], false, false);
    }
  }
  #pragma unroll
  for (int nt = 0; nt < 8; ++nt) {
    int n = nt*16 + lo;
    float bias = besh[n];
    #pragma unroll
    for (int v = 0; v < 8; ++v) {
      int m = v + 8*hi;                                    // C layout: M = v + 8*hi, N = lo
      he[(eb + m)*Hh + n] = bf_bits((__bf16)(c[nt][v] + bias));
    }
  }
}

// ---------------- big WMMA kernel 2: layer-2 edge scores ----------------
// s2_e = leaky( xl2[i] + xr2[dst] + mask_i*(h_e@W2a + ma_e*P2[S[dst]]) ) . g2_att
__global__ __launch_bounds__(256) void k_gat2_score(
    const unsigned short* __restrict__ he, const int* __restrict__ E_idx,
    const int* __restrict__ S, const float* __restrict__ mask,
    const float* __restrict__ g2_We, const float* __restrict__ g2_att,
    const float* __restrict__ xl2A, const float* __restrict__ xl2B,
    const float* __restrict__ xr2A, const float* __restrict__ xr2B,
    const float* __restrict__ P2, const int* __restrict__ inv, const int* __restrict__ occ,
    float* __restrict__ s2, unsigned* __restrict__ m2enc) {
  __shared__ __align__(32) __bf16 Wsh[Hh*Hh];   // W2a = g2_We[0:H] transposed [n][k]
  __shared__ float P2sh[Vv*Hh];
  __shared__ float xlA[Hh], xlB[Hh], xrA[Hh], xrB[Hh], attsh[Hh];
  int tid = threadIdx.x;
  for (int idx = tid; idx < Hh*Hh; idx += 256) {
    int k = idx / Hh, n = idx % Hh;
    Wsh[n*Hh + k] = (__bf16)g2_We[k*Hh + n];
  }
  for (int idx = tid; idx < Vv*Hh; idx += 256) P2sh[idx] = P2[idx];
  if (tid < Hh) {
    xlA[tid] = xl2A[tid]; xlB[tid] = xl2B[tid];
    xrA[tid] = xr2A[tid]; xrB[tid] = xr2B[tid];
    attsh[tid] = g2_att[tid];
  }
  __syncthreads();

  int wave = tid >> 5, lane = tid & 31;
  int hi = lane >> 4, lo = lane & 15;
  size_t eb = ((size_t)blockIdx.x * 8 + wave) * 16;
  int i = (int)(eb / Kk);        // 16-edge tile lies within one source node (K=32)
  float mi = mask[i];
  int invi = inv[i];
  const float* xls = occ[i] ? xlB : xlA;

  int dstv[8], occd[8], srw[8];
  float mav[8];
  #pragma unroll
  for (int v = 0; v < 8; ++v) {
    int m = v + 8*hi;
    int d = E_idx[eb + m];
    dstv[v] = d;
    mav[v]  = (invi > inv[d]) ? 1.f : 0.f;
    occd[v] = occ[d];
    srw[v]  = S[d];
  }

  const unsigned short* Arow = he + (eb + lo) * Hh;
  v8f c[8] = {};
  for (int kt = 0; kt < 4; ++kt) {
    v16bf a;
    #pragma unroll
    for (int r = 0; r < 8; ++r) {
      int e = 2*r;
      int k = 16*hi + e;
      unsigned u = *(const unsigned*)(Arow + kt*32 + k);
      a[e]   = bits_bf((unsigned short)(u & 0xffffu));
      a[e+1] = bits_bf((unsigned short)(u >> 16));
    }
    #pragma unroll
    for (int nt = 0; nt < 8; ++nt) {
      int n = nt*16 + lo;
      v16bf b = *(const v16bf*)&Wsh[n*Hh + kt*32 + hi*16];
      c[nt] = __builtin_amdgcn_wmma_f32_16x16x32_bf16(false, a, false, b, (short)0, c[nt], false, false);
    }
  }

  float p[8] = {0.f,0.f,0.f,0.f,0.f,0.f,0.f,0.f};
  #pragma unroll
  for (int nt = 0; nt < 8; ++nt) {
    int n = nt*16 + lo;
    float an  = attsh[n];
    float xln = xls[n];
    #pragma unroll
    for (int v = 0; v < 8; ++v) {
      float z = xln + (occd[v] ? xrB[n] : xrA[n])
              + mi * (c[nt][v] + mav[v] * P2sh[srw[v]*Hh + n]);
      float l = (z > 0.f) ? z : NEGS*z;
      p[v] += l * an;
    }
  }
  #pragma unroll
  for (int off = 1; off < 16; off <<= 1) {
    #pragma unroll
    for (int v = 0; v < 8; ++v) p[v] += __shfl_xor(p[v], off);
  }
  if (lo == 0) {
    #pragma unroll
    for (int v = 0; v < 8; ++v) {
      int m = v + 8*hi;
      s2[eb + m] = p[v];
      atomicMax(&m2enc[dstv[v]], fenc(p[v]));
    }
  }
}

// ---------------- softmax passes + output ----------------

__global__ void k_expw(const float* __restrict__ s2, const int* __restrict__ E_idx,
                       const unsigned* __restrict__ m2enc, float* __restrict__ wbuf,
                       float* __restrict__ d2) {
  int e = blockIdx.x*blockDim.x + threadIdx.x;
  if (e >= NK) return;
  int d = E_idx[e];
  float w = expf(s2[e] - fdec(m2enc[d]));
  wbuf[e] = w;
  atomicAdd(&d2[d], w);
}

__global__ void k_alpha(const float* __restrict__ wbuf, const int* __restrict__ E_idx,
                        const float* __restrict__ d2, const int* __restrict__ occ,
                        float* __restrict__ SA, float* __restrict__ SB) {
  int e = blockIdx.x*blockDim.x + threadIdx.x;
  if (e >= NK) return;
  int d = E_idx[e];
  float alpha = wbuf[e] / (d2[d] + 1e-16f);
  int i = e / Kk;   // src node
  atomicAdd(occ[i] ? &SB[d] : &SA[d], alpha);
}

__global__ void k_logits(const float* __restrict__ SA, const float* __restrict__ SB,
                         const float* __restrict__ pA, const float* __restrict__ pB,
                         const float* __restrict__ pbase, float* __restrict__ out) {
  int d = blockIdx.x*blockDim.x + threadIdx.x;
  if (d >= Nn) return;
  float sa = SA[d], sb = SB[d];
  float l[Vv];
  float m = -1e30f;
  for (int v = 0; v < Vv; ++v) { l[v] = pbase[v] + sa*pA[v] + sb*pB[v]; m = fmaxf(m, l[v]); }
  float s = 0.f;
  for (int v = 0; v < Vv; ++v) s += expf(l[v] - m);
  float lse = m + logf(s);
  for (int v = 0; v < Vv; ++v) out[d*Vv + v] = l[v] - lse;
}

// ---------------- host launcher ----------------

extern "C" void kernel_launch(void* const* d_in, const int* in_sizes, int n_in,
                              void* d_out, int out_size, void* d_ws, size_t ws_size,
                              hipStream_t stream) {
  (void)in_sizes; (void)n_in; (void)out_size; (void)ws_size;
  const float* E       = (const float*)d_in[0];
  const int*   E_idx   = (const int*)  d_in[1];
  const int*   S       = (const int*)  d_in[2];
  const float* mask    = (const float*)d_in[3];
  const float* chain_M = (const float*)d_in[4];
  const float* noise   = (const float*)d_in[5];
  const float* W_e     = (const float*)d_in[6];
  const float* b_e     = (const float*)d_in[7];
  const float* W_s     = (const float*)d_in[8];
  const float* g1_bl   = (const float*)d_in[10];
  const float* g1_b    = (const float*)d_in[15];
  const float* g2_Wl   = (const float*)d_in[16];
  const float* g2_bl   = (const float*)d_in[17];
  const float* g2_Wr   = (const float*)d_in[18];
  const float* g2_br   = (const float*)d_in[19];
  const float* g2_We   = (const float*)d_in[20];
  const float* g2_att  = (const float*)d_in[21];
  const float* g2_b    = (const float*)d_in[22];
  const float* W_out   = (const float*)d_in[23];
  const float* b_out   = (const float*)d_in[24];
  float* out = (float*)d_out;

  // workspace carve (256B aligned)
  size_t off = 0;
  auto carve = [&](size_t bytes) {
    void* p = (char*)d_ws + off;
    off += (bytes + 255) & ~(size_t)255;
    return p;
  };
  unsigned short* he   = (unsigned short*)carve((size_t)NK * Hh * 2);
  float*    s2    = (float*)   carve((size_t)NK * 4);
  float*    wbuf  = (float*)   carve((size_t)NK * 4);
  float*    keys  = (float*)   carve((size_t)Nn * 4);
  int*      inv   = (int*)     carve((size_t)Nn * 4);
  int*      occ   = (int*)     carve((size_t)Nn * 4);
  unsigned* m2    = (unsigned*)carve((size_t)Nn * 4);
  float*    d2    = (float*)   carve((size_t)Nn * 4);
  float*    SA    = (float*)   carve((size_t)Nn * 4);
  float*    SB    = (float*)   carve((size_t)Nn * 4);
  float*    xl2A  = (float*)   carve(Hh * 4);
  float*    xl2B  = (float*)   carve(Hh * 4);
  float*    xr2A  = (float*)   carve(Hh * 4);
  float*    xr2B  = (float*)   carve(Hh * 4);
  float*    P2    = (float*)   carve((size_t)Vv * Hh * 4);
  float*    pA    = (float*)   carve(32 * 4);
  float*    pB    = (float*)   carve(32 * 4);
  float*    pbase = (float*)   carve(32 * 4);

  k_keys<<<Nn/256, 256, 0, stream>>>(chain_M, mask, noise, keys);
  k_rank<<<Nn/256, 256, 0, stream>>>(keys, inv);
  k_init<<<Nn/256, 256, 0, stream>>>(m2, d2, SA, SB, occ);
  k_occ<<<NK/256, 256, 0, stream>>>(E_idx, occ);
  k_prep<<<1, 256, 0, stream>>>(W_s, g1_bl, g1_b, g2_Wl, g2_bl, g2_Wr, g2_br,
                                g2_We, g2_b, W_out, b_out,
                                xl2A, xl2B, xr2A, xr2B, P2, pA, pB, pbase);
  // 2048 blocks x 8 waves x 16 edges = 262144 edges
  k_he_gemm<<<NK/128, 256, 0, stream>>>(E, W_e, b_e, he);
  k_gat2_score<<<NK/128, 256, 0, stream>>>(he, E_idx, S, mask, g2_We, g2_att,
                                           xl2A, xl2B, xr2A, xr2B, P2, inv, occ, s2, m2);
  k_expw<<<NK/256, 256, 0, stream>>>(s2, E_idx, m2, wbuf, d2);
  k_alpha<<<NK/256, 256, 0, stream>>>(wbuf, E_idx, d2, occ, SA, SB);
  k_logits<<<Nn/256, 256, 0, stream>>>(SA, SB, pA, pB, pbase, out);
}
